// WUDT_STAnet_18708877541723
// MI455X (gfx1250) — matrix operand order repaired and checked by reading, not compiled, not run on hardware
//
#include <hip/hip_runtime.h>
#include <hip/hip_bf16.h>

typedef __attribute__((ext_vector_type(16))) _Float16 v16h;
typedef __attribute__((ext_vector_type(8)))  float    v8f;
typedef long long ll;

#define CDIV(a,b) (((a)+(b)-1)/(b))

// ---------------------------------------------------------------------------
// Elementwise / repack kernels
// ---------------------------------------------------------------------------

__global__ void k_nchw_to_nlc(const float* __restrict__ x, float* __restrict__ h, int total) {
  int i = blockIdx.x * 256 + threadIdx.x;
  if (i >= total) return;
  int l = i & 4095; int r = i >> 12; int c = r & 255; int b = r >> 8;
  h[((ll)(b * 4096 + l)) * 256 + c] = x[i];
}

__global__ void k_cvt_f16(const float* __restrict__ a, _Float16* __restrict__ o, int n) {
  int i = blockIdx.x * 256 + threadIdx.x;
  if (i < n) o[i] = (_Float16)a[i];
}

// LayerNorm over C=256 (one wave per token) + optional window rearrange, f16 out.
__global__ void k_ln(const float* __restrict__ x, const float* __restrict__ g,
                     const float* __restrict__ bt, _Float16* __restrict__ out,
                     int wh, int ww) {
  int tok  = blockIdx.x * 8 + (threadIdx.x >> 5);
  int lane = threadIdx.x & 31;
  int b = tok >> 12, l = tok & 4095;
  const float* xp = x + (ll)tok * 256;
  float v[8], s = 0.f, ss = 0.f;
#pragma unroll
  for (int i = 0; i < 8; ++i) { float t = xp[lane + i * 32]; v[i] = t; s += t; ss += t * t; }
#pragma unroll
  for (int m = 16; m; m >>= 1) { s += __shfl_xor(s, m, 32); ss += __shfl_xor(ss, m, 32); }
  float mu  = s * (1.f / 256.f);
  float var = ss * (1.f / 256.f) - mu * mu;
  float rs  = rsqrtf(var + 1e-5f);
  ll ot;
  if (wh) {
    int y = l >> 6, xc = l & 63;
    int Nw = (64 / wh) * (64 / ww);
    int n = (y / wh) * (64 / ww) + xc / ww;
    int sidx = (y % wh) * ww + (xc % ww);
    ot = ((ll)(b * 4096 + sidx * Nw + n)) * 256;
  } else {
    ot = (ll)tok * 256;
  }
#pragma unroll
  for (int i = 0; i < 8; ++i) {
    int c = lane + i * 32;
    out[ot + c] = (_Float16)((v[i] - mu) * rs * g[c] + bt[c]);
  }
}

// qkv f32 (B, S*N, 3C) -> q,k f16 (B, H=8, S, d) following the reference's
// reshape(B,S,3,H,N*C/H) reinterpretation (q/k/v scrambled across windows).
__global__ void k_qk_repack(const float* __restrict__ qkv, _Float16* __restrict__ q,
                            _Float16* __restrict__ k,
                            int S, int N, int d, int total) {
  int i = blockIdx.x * 256 + threadIdx.x;
  if (i >= total) return;
  int dd = i % d; int r = i / d; int s = r % S; r /= S; int h = r & 7; int b = r >> 3;
#pragma unroll
  for (int sel = 0; sel < 2; ++sel) {
    int f = sel * 8 * d + h * d + dd;
    int n = f / 768; int j = f - n * 768;
    _Float16 hv = (_Float16)qkv[((ll)(b * 4096 + s * N + n)) * 768 + j];
    if (sel == 0) q[i] = hv; else k[i] = hv;
  }
}

// qkv f32 -> V TRANSPOSED f16 (B, H, d, S) row-major, so that attn@V can run
// the BT=true GEMM path (contiguous 32B B-fragment loads). Output-coalesced.
__global__ void k_vT_repack(const float* __restrict__ qkv, _Float16* __restrict__ vT,
                            int S, int N, int d, int total) {
  int i = blockIdx.x * 256 + threadIdx.x;
  if (i >= total) return;
  int s = i % S; int r = i / S; int dd = r % d; r /= d; int h = r & 7; int b = r >> 3;
  int f = 16 * d + h * d + dd;           // sel = 2 (V)
  int n = f / 768; int j = f - n * 768;
  vT[i] = (_Float16)qkv[((ll)(b * 4096 + s * N + n)) * 768 + j];
}

// ctx f32 (B,H,S,d) -> proj input f16 (B, S*N, C)  (transpose(0,2,1,3).reshape)
__global__ void k_ctx_repack(const float* __restrict__ ctx, _Float16* __restrict__ out,
                             int S, int N, int d, int total) {
  int i = blockIdx.x * 256 + threadIdx.x;
  if (i >= total) return;
  int c = i & 255; int r = i >> 8; int t = r & 4095; int b = r >> 12;
  int s = t / N; int n = t - s * N;
  int g = n * 256 + c; int h = g / d; int dd = g - h * d;
  out[i] = (_Float16)ctx[((ll)((b * 8 + h) * S + s)) * d + dd];
}

// h += window_reverse(proj_out)
__global__ void k_win_rev_add(float* h, const float* __restrict__ p,
                              int wh, int ww, int total) {
  int i = blockIdx.x * 256 + threadIdx.x;
  if (i >= total) return;
  int c = i & 255; int r = i >> 8; int l = r & 4095; int b = r >> 12;
  int y = l >> 6, x = l & 63;
  int Nw = (64 / wh) * (64 / ww);
  int n = (y / wh) * (64 / ww) + x / ww;
  int s = (y % wh) * ww + (x % ww);
  h[i] += p[((ll)(b * 4096 + s * Nw + n)) * 256 + c];
}

// softmax(scale*scores + rel-pos bias) -> f16, one wave per row
__global__ void k_softmax(const float* __restrict__ sc, const float* __restrict__ table,
                          _Float16* __restrict__ out, int S, int wh, int ww, float scale) {
  int row  = blockIdx.x * 8 + (threadIdx.x >> 5);
  int lane = threadIdx.x & 31;
  int h = (row / S) & 7;
  int s = row % S;
  int si = s / ww, sj = s - si * ww;
  const float* rp = sc + (ll)row * S;
  float tmp[16];
  float mx = -1e30f;
  int nt = S >> 5;
  for (int it = 0; it < nt; ++it) {
    int t = lane + (it << 5);
    int ti = t / ww, tj = t - ti * ww;
    int idx = (si - ti + wh - 1) * (2 * ww - 1) + (sj - tj + ww - 1);
    float v = rp[t] * scale + table[idx * 8 + h];
    tmp[it] = v; mx = fmaxf(mx, v);
  }
#pragma unroll
  for (int m = 16; m; m >>= 1) mx = fmaxf(mx, __shfl_xor(mx, m, 32));
  float sum = 0.f;
  for (int it = 0; it < nt; ++it) { tmp[it] = __expf(tmp[it] - mx); sum += tmp[it]; }
#pragma unroll
  for (int m = 16; m; m >>= 1) sum += __shfl_xor(sum, m, 32);
  float inv = 1.f / sum;
  _Float16* op = out + (ll)row * S;
  for (int it = 0; it < nt; ++it) op[lane + (it << 5)] = (_Float16)(tmp[it] * inv);
}

// depthwise 3x3 (SAME) + bias + exact GELU, (B,L,1024) f32 -> f16
__global__ void k_dwconv_gelu(const float* __restrict__ hid, const float* __restrict__ w,
                              const float* __restrict__ bias, _Float16* __restrict__ out,
                              int total) {
  int i = blockIdx.x * 256 + threadIdx.x;
  if (i >= total) return;
  int ch = i & 1023; int r = i >> 10; int l = r & 4095; int b = r >> 12;
  int y = l >> 6, x = l & 63;
  float acc = bias[ch];
#pragma unroll
  for (int dy = 0; dy < 3; ++dy)
#pragma unroll
    for (int dx = 0; dx < 3; ++dx) {
      int yy = y + dy - 1, xx = x + dx - 1;
      if (yy >= 0 && yy < 64 && xx >= 0 && xx < 64)
        acc += hid[((ll)(b * 4096 + yy * 64 + xx)) * 1024 + ch] * w[ch * 9 + dy * 3 + dx];
    }
  out[i] = (_Float16)(0.5f * acc * (1.f + erff(acc * 0.70710678118654752f)));
}

// (B,L,C) f32 -> zero-padded (B,66,66,C) f16 for implicit-GEMM conv
__global__ void k_pad_f16(const float* __restrict__ h, _Float16* __restrict__ o, int total) {
  int i = blockIdx.x * 256 + threadIdx.x;
  if (i >= total) return;
  int c = i & 255; int r = i >> 8;
  int x = r % 66; r /= 66; int y = r % 66; int b = r / 66;
  float v = 0.f;
  if (y >= 1 && y <= 64 && x >= 1 && x <= 64)
    v = h[((ll)(b * 4096 + (y - 1) * 64 + (x - 1))) * 256 + c];
  o[i] = (_Float16)v;
}

// conv_w (O,I,3,3) f32 -> per-tap (t,O,I) f16 weight matrices (N x K, row-major)
__global__ void k_convw_repack(const float* __restrict__ w, _Float16* __restrict__ o, int total) {
  int i = blockIdx.x * 256 + threadIdx.x;
  if (i >= total) return;
  int ci = i & 255; int r = i >> 8; int oc = r & 255; int t = r >> 8;
  o[i] = (_Float16)w[(oc * 256 + ci) * 9 + t];
}

// (B,L,C) f32 -> NCHW f32 output with LeakyReLU(0.1)
__global__ void k_out_leaky(const float* __restrict__ cin, float* __restrict__ out, int total) {
  int i = blockIdx.x * 256 + threadIdx.x;
  if (i >= total) return;
  int l = i & 4095; int r = i >> 12; int c = r & 255; int b = r >> 8;
  float v = cin[((ll)(b * 4096 + l)) * 256 + c];
  out[i] = v > 0.f ? v : 0.1f * v;
}

// ---------------------------------------------------------------------------
// WMMA GEMM:  C[M,N] f32 = A[M,K] f16  x  W[N,K] f16 (+ bias + residual)
//   CONVA : A is a padded (B,66,66,256) f16 image, implicit im2col via convOff
// A-tiles are moved global->LDS with the CDNA5 async DMA path
// (global_load_async_to_lds_b128, ASYNCcnt), double-buffered so the copy of
// tile kt+1 overlaps the WMMAs on tile kt. One s_wait_asynccnt + one barrier
// per K-step. Each wave computes a 32x32 tile (2x2 v_wmma_f32_16x16x32_f16).
// B fragments are single contiguous 32B v16h loads (all operands pre-packed
// as N x K row-major, including transposed V for attn@V).
// ---------------------------------------------------------------------------
template <int BM, int BN, bool CONVA>
__global__ __launch_bounds__(BM * BN / 32)
void gemm_wmma(const _Float16* __restrict__ A, const _Float16* __restrict__ Bm,
               const float* __restrict__ bias, const float* resid, float* Cm,
               int M, int N, int K, ll sA, ll sB, ll sC, int convOff) {
  constexpr int WM  = BM / 32;
  constexpr int NT  = BM * BN / 32;  // threads
  constexpr int LDA = 40;            // padded LDS stride (halves)
  constexpr int BUF = BM * LDA;      // halves per buffer
  constexpr int CH  = (BM * 4) / NT; // 16B async chunks per thread per tile
  __shared__ _Float16 As[2 * BUF];

  int bz = blockIdx.z;
  if (!CONVA) A += (ll)bz * sA;
  Bm += (ll)bz * sB;
  Cm += (ll)bz * sC;
  if (resid) resid += (ll)bz * sC;

  int tid = threadIdx.x, lane = tid & 31, wid = tid >> 5;
  int wm = wid % WM, wn = wid / WM;
  int m0 = blockIdx.y * BM, n0 = blockIdx.x * BN;

  // Per-thread A-tile chunk descriptors (one b128 per chunk).
  const _Float16* srcBase[CH];
  unsigned ldsOff[CH];
#pragma unroll
  for (int c = 0; c < CH; ++c) {
    int idx = tid + c * NT;
    int r = idx >> 2, kk = (idx & 3) << 3;
    if (CONVA) {
      int gm = m0 + r;
      int b = gm >> 12, y = (gm >> 6) & 63, x = gm & 63;
      srcBase[c] = A + ((ll)((b * 66 + y) * 66 + x)) * K + convOff + kk;
    } else {
      srcBase[c] = A + (ll)(m0 + r) * K + kk;
    }
    ldsOff[c] = (unsigned)(size_t)&As[r * LDA + kk];
  }

  const int nKT = K >> 5;
  // prologue: async DMA of tile 0 into buffer 0
#pragma unroll
  for (int c = 0; c < CH; ++c)
    asm volatile("global_load_async_to_lds_b128 %0, %1, off"
                 :: "v"(ldsOff[c]), "v"(srcBase[c]) : "memory");

  v8f acc[2][2] = {};
  for (int kt = 0; kt < nKT; ++kt) {
    // my DMA for tile kt has landed; barrier makes all waves' writes visible
    // and guarantees nobody still reads the buffer the next DMA will fill.
    asm volatile("s_wait_asynccnt 0x0" ::: "memory");
    __syncthreads();
    if (kt + 1 < nKT) {
      unsigned bufo = (unsigned)(((kt + 1) & 1) * (BUF * 2));  // byte offset
#pragma unroll
      for (int c = 0; c < CH; ++c)
        asm volatile("global_load_async_to_lds_b128 %0, %1, off"
                     :: "v"(ldsOff[c] + bufo), "v"(srcBase[c] + (kt + 1) * 32) : "memory");
    }

    const _Float16* cur = &As[(kt & 1) * BUF];
    // A fragments: lanes 0-15 K0..15, lanes 16-31 K16..31 (matches B)
    v16h a[2], b[2];
    int kb16 = (lane >> 4) << 4;
#pragma unroll
    for (int i = 0; i < 2; ++i) {
      const _Float16* p = cur + (wm * 32 + i * 16 + (lane & 15)) * LDA;
#pragma unroll
      for (int q = 0; q < 8; ++q) a[i][q] = p[kb16 + q];
#pragma unroll
      for (int q = 0; q < 8; ++q) a[i][8 + q] = p[kb16 + 8 + q];
    }
    // B fragments, same contiguous k order as A: lanes 0-15 K0..15, lanes 16-31 K16..31 of column n

#pragma unroll
    for (int j = 0; j < 2; ++j) {
      int n = n0 + wn * 32 + j * 16 + (lane & 15);
      const _Float16* bp = Bm + (ll)n * K + kt * 32 + kb16;
      b[j] = *(const v16h*)bp;  // 32B contiguous
      // branch-free speculative prefetch of next K-step (clamped address)
      __builtin_prefetch(bp + ((kt + 1 < nKT) ? 32 : 0), 0, 0);
    }
#pragma unroll
    for (int i = 0; i < 2; ++i)
#pragma unroll
      for (int j = 0; j < 2; ++j)
        acc[i][j] = __builtin_amdgcn_wmma_f32_16x16x32_f16(
            false, a[i], false, b[j], (short)0, acc[i][j], false, false);
  }

  // epilogue: C/D layout -> m = base + r (+8 for lanes 16-31), n = base + (lane&15)
#pragma unroll
  for (int i = 0; i < 2; ++i) {
    int mBase = m0 + wm * 32 + i * 16 + ((lane >> 4) << 3);
#pragma unroll
    for (int j = 0; j < 2; ++j) {
      int n = n0 + wn * 32 + j * 16 + (lane & 15);
      float bv = bias ? bias[n] : 0.f;
#pragma unroll
      for (int r = 0; r < 8; ++r) {
        ll o = (ll)(mBase + r) * N + n;
        float vv = acc[i][j][r] + bv;
        if (resid) vv += resid[o];
        Cm[o] = vv;
      }
    }
  }
}

// ---------------------------------------------------------------------------
// Host orchestration
// ---------------------------------------------------------------------------
extern "C" void kernel_launch(void* const* d_in, const int* in_sizes, int n_in,
                              void* d_out, int out_size, void* d_ws, size_t ws_size,
                              hipStream_t stream) {
  (void)in_sizes; (void)n_in; (void)out_size; (void)ws_size;
  const float* x      = (const float*)d_in[0];
  const float* conv_w = (const float*)d_in[91];

  const size_t MB = 1ull << 20;
  char* ws = (char*)d_ws;
  float*    hbuf   = (float*)ws;                          // 16 MB residual stream (B,L,C)
  _Float16* w16a   = (_Float16*)(ws + 16 * MB);           // qkv_w f16 / conv taps f16
  _Float16* w16b   = (_Float16*)(ws + 17 * MB);           // proj_w f16
  _Float16* w16c   = (_Float16*)(ws + 18 * MB);           // fc1_w f16
  _Float16* w16d   = (_Float16*)(ws + 19 * MB);           // fc2_w f16
  float*    bigf   = (float*)(ws + 20 * MB);              // 64 MB: qkv / proj / fc1 out
  _Float16* a16    = (_Float16*)(ws + 84 * MB);           // 8 MB GEMM-A staging
  _Float16* q16    = (_Float16*)(ws + 92 * MB);           // 8 MB each
  _Float16* k16    = (_Float16*)(ws + 100 * MB);
  _Float16* vT16   = (_Float16*)(ws + 108 * MB);          // V transposed (B,H,d,S)
  float*    scores = (float*)(ws + 116 * MB);             // 32 MB (also gelu16)
  _Float16* gelu16 = (_Float16*)(ws + 116 * MB);
  _Float16* attn16 = (_Float16*)(ws + 148 * MB);          // 16 MB
  float*    ctx    = (float*)(ws + 164 * MB);             // 16 MB (also conv acc)
  _Float16* pad16  = (_Float16*)(ws + 92 * MB);           // reuse q16 region at the end

  k_nchw_to_nlc<<<16384, 256, 0, stream>>>(x, hbuf, 4194304);

  static const int WH[6] = {64, 8, 8, 64, 8, 8};
  static const int WW[6] = { 8, 64, 8,  8, 64, 8};
  const float SCALE = 0.17677669529663687f;  // (C/HEADS)^-0.5 = 32^-0.5

  for (int blk = 0; blk < 6; ++blk) {
    const float* n1g  = (const float*)d_in[1 + blk * 15 + 0];
    const float* n1b  = (const float*)d_in[1 + blk * 15 + 1];
    const float* qkvw = (const float*)d_in[1 + blk * 15 + 2];
    const float* qkvb = (const float*)d_in[1 + blk * 15 + 3];
    const float* pw   = (const float*)d_in[1 + blk * 15 + 4];
    const float* pb   = (const float*)d_in[1 + blk * 15 + 5];
    const float* btab = (const float*)d_in[1 + blk * 15 + 6];
    const float* n2g  = (const float*)d_in[1 + blk * 15 + 7];
    const float* n2b  = (const float*)d_in[1 + blk * 15 + 8];
    const float* f1w  = (const float*)d_in[1 + blk * 15 + 9];
    const float* f1b  = (const float*)d_in[1 + blk * 15 + 10];
    const float* dww  = (const float*)d_in[1 + blk * 15 + 11];
    const float* dwb  = (const float*)d_in[1 + blk * 15 + 12];
    const float* f2w  = (const float*)d_in[1 + blk * 15 + 13];
    const float* f2b  = (const float*)d_in[1 + blk * 15 + 14];

    int wh = WH[blk], ww = WW[blk];
    int S = wh * ww, N = 4096 / S, d = N * 32;

    // --- attention half ---
    k_ln<<<2048, 256, 0, stream>>>(hbuf, n1g, n1b, a16, wh, ww);
    k_cvt_f16<<<CDIV(768 * 256, 256), 256, 0, stream>>>(qkvw, w16a, 768 * 256);
    k_cvt_f16<<<CDIV(256 * 256, 256), 256, 0, stream>>>(pw, w16b, 256 * 256);
    k_cvt_f16<<<CDIV(1024 * 256, 256), 256, 0, stream>>>(f1w, w16c, 1024 * 256);
    k_cvt_f16<<<CDIV(256 * 1024, 256), 256, 0, stream>>>(f2w, w16d, 256 * 1024);

    gemm_wmma<64, 128, false><<<dim3(6, 256, 1), 256, 0, stream>>>(
        a16, w16a, qkvb, nullptr, bigf, 16384, 768, 256, 0, 0, 0, 0);
    k_qk_repack<<<16384, 256, 0, stream>>>(bigf, q16, k16, S, N, d, 4194304);
    k_vT_repack<<<16384, 256, 0, stream>>>(bigf, vT16, S, N, d, 4194304);

    if (S == 512) {
      gemm_wmma<64, 128, false><<<dim3(4, 8, 32), 256, 0, stream>>>(
          q16, k16, nullptr, nullptr, scores, 512, 512, 256,
          (ll)512 * 256, (ll)512 * 256, (ll)512 * 512, 0);
    } else {
      gemm_wmma<64, 64, false><<<dim3(1, 1, 32), 128, 0, stream>>>(
          q16, k16, nullptr, nullptr, scores, 64, 64, 2048,
          (ll)64 * 2048, (ll)64 * 2048, (ll)64 * 64, 0);
    }
    k_softmax<<<4 * S, 256, 0, stream>>>(scores, btab, attn16, S, wh, ww, SCALE);
    // attn @ V with W = vT (N=d rows, K=S cols, row-major) -> contiguous B loads
    if (S == 512) {
      gemm_wmma<64, 128, false><<<dim3(2, 8, 32), 256, 0, stream>>>(
          attn16, vT16, nullptr, nullptr, ctx, 512, 256, 512,
          (ll)512 * 512, (ll)256 * 512, (ll)512 * 256, 0);
    } else {
      gemm_wmma<64, 128, false><<<dim3(16, 1, 32), 256, 0, stream>>>(
          attn16, vT16, nullptr, nullptr, ctx, 64, 2048, 64,
          (ll)64 * 64, (ll)2048 * 64, (ll)64 * 2048, 0);
    }
    k_ctx_repack<<<16384, 256, 0, stream>>>(ctx, a16, S, N, d, 4194304);
    gemm_wmma<64, 128, false><<<dim3(2, 256, 1), 256, 0, stream>>>(
        a16, w16b, pb, nullptr, bigf, 16384, 256, 256, 0, 0, 0, 0);
    k_win_rev_add<<<16384, 256, 0, stream>>>(hbuf, bigf, wh, ww, 4194304);

    // --- MLP half ---
    k_ln<<<2048, 256, 0, stream>>>(hbuf, n2g, n2b, a16, 0, 0);
    gemm_wmma<64, 128, false><<<dim3(8, 256, 1), 256, 0, stream>>>(
        a16, w16c, f1b, nullptr, bigf, 16384, 1024, 256, 0, 0, 0, 0);
    k_dwconv_gelu<<<65536, 256, 0, stream>>>(bigf, dww, dwb, gelu16, 16777216);
    gemm_wmma<64, 128, false><<<dim3(2, 256, 1), 256, 0, stream>>>(
        gelu16, w16d, f2b, hbuf, hbuf, 16384, 256, 1024, 0, 0, 0, 0);
  }

  // --- final 3x3 conv as 9 accumulating implicit-GEMM taps + LeakyReLU ---
  k_pad_f16<<<CDIV(4460544, 256), 256, 0, stream>>>(hbuf, pad16, 4460544);
  k_convw_repack<<<CDIV(589824, 256), 256, 0, stream>>>(conv_w, w16a, 589824);
  for (int t = 0; t < 9; ++t) {
    int dy = t / 3, dx = t % 3;
    gemm_wmma<64, 128, true><<<dim3(2, 256, 1), 256, 0, stream>>>(
        pad16, w16a + t * 65536, nullptr, (t == 0 ? nullptr : (const float*)ctx), ctx,
        16384, 256, 256, 0, 0, 0, (dy * 66 + dx) * 256);
  }
  k_out_leaky<<<16384, 256, 0, stream>>>(ctx, (float*)d_out, 4194304);
}
